// HSRCompressorLLaVA_81097572483612
// MI455X (gfx1250) — matrix operand order repaired and checked by reading, not compiled, not run on hardware
//
#include <hip/hip_runtime.h>
#include <hip/hip_bf16.h>
#include <math.h>

// ---------------- problem constants ----------------
#define BB   16
#define NN   2304
#define DD   1024
#define SIDE 48
#define MM   1728      // N - K_anchor
#define KK   576       // K_cent
#define KA   576       // K_anchor
#define NF   1152      // Ka + K
#define SW   0.1f      // spatial_weight
#define NITER 10

// ---------------- vector types ----------------
typedef __attribute__((ext_vector_type(8)))  float          f32x8;
typedef __attribute__((ext_vector_type(8)))  unsigned short u16x8;
typedef __attribute__((ext_vector_type(16))) unsigned short u16x16;
typedef __attribute__((ext_vector_type(16))) __bf16         bf16x16;

__device__ __forceinline__ unsigned short f2bf(float f) {
    unsigned u = __builtin_bit_cast(unsigned, f);
    u += 0x7FFFu + ((u >> 16) & 1u);          // round-to-nearest-even
    return (unsigned short)(u >> 16);
}

__device__ __forceinline__ bf16x16 mk_frag(u16x8 lo, u16x8 hi) {
    u16x16 v = __builtin_shufflevector(lo, hi, 0,1,2,3,4,5,6,7,8,9,10,11,12,13,14,15);
    return __builtin_bit_cast(bf16x16, v);
}

// A fragment: 16x32 bf16, row-major source `rowp` points at (row, d0).
// lane<16: K 0..15; lane>=16: K 16..31 (contiguous, matches B)
__device__ __forceinline__ bf16x16 ld_fragA(const unsigned short* rowp, int h) {
    u16x8 lo = *(const u16x8*)(rowp + h * 16);
    u16x8 hi = *(const u16x8*)(rowp + h * 16 + 8);
    return mk_frag(lo, hi);
}
// B fragment: 32x16 bf16 (row `col` of C, contiguous d). lane half selects K 0..15 / 16..31.
__device__ __forceinline__ bf16x16 ld_fragB(const unsigned short* p) {
    u16x8 lo = *(const u16x8*)(p);
    u16x8 hi = *(const u16x8*)(p + 8);
    return mk_frag(lo, hi);
}

// ---- CDNA5 async global->LDS DMA (ASYNCcnt-tracked, bypasses VGPRs) ----
// Low 32 bits of a generic pointer into the LDS aperture are the raw LDS
// byte address (ISA 10.2). IOFFSET applies to both
// the LDS and the global address, so the 32-byte transfer shares one base pair.
__device__ __forceinline__ void async_b128x2(const unsigned short* g, unsigned short* l) {
    unsigned lds = (unsigned)(unsigned long long)(uintptr_t)l;
    unsigned long long ga = (unsigned long long)(uintptr_t)g;
    asm volatile("global_load_async_to_lds_b128 %0, %1, off\n\t"
                 "global_load_async_to_lds_b128 %0, %1, off offset:16"
                 :: "v"(lds), "v"(ga) : "memory");
}
__device__ __forceinline__ void wait_async0() {
    asm volatile("s_wait_asynccnt 0x0" ::: "memory");
}

// ---------------- small utility kernels ----------------
__global__ void k_zero_f32(float* __restrict__ p, long n) {
    long i = (long)blockIdx.x * blockDim.x + threadIdx.x;
    if (i < n) p[i] = 0.0f;
}

__global__ __launch_bounds__(256) void k_norm2(const float* __restrict__ x, float* __restrict__ norm2) {
    __shared__ float red[256];
    long t = blockIdx.x;                     // b*NN + n
    const float* row = x + t * DD;
    float s = 0.f;
    for (int d = threadIdx.x; d < DD; d += 256) { float v = row[d]; s += v * v; }
    red[threadIdx.x] = s; __syncthreads();
    for (int o = 128; o > 0; o >>= 1) { if (threadIdx.x < o) red[threadIdx.x] += red[threadIdx.x + o]; __syncthreads(); }
    if (threadIdx.x == 0) norm2[t] = red[0];
}

// O(N^2) rank per batch -> top_idx in descending-saliency order (tie: lower idx)
__global__ __launch_bounds__(256) void k_rank(const float* __restrict__ norm2,
                                              int* __restrict__ rank, int* __restrict__ top_idx) {
    __shared__ float s[NN];
    int b = blockIdx.x;
    for (int i = threadIdx.x; i < NN; i += 256) s[i] = norm2[b * NN + i];
    __syncthreads();
    for (int i = threadIdx.x; i < NN; i += 256) {
        float v = s[i]; int r = 0;
        for (int j = 0; j < NN; ++j) { float u = s[j]; r += (u > v) || (u == v && j < i); }
        rank[b * NN + i] = r;
        if (r < KA) top_idx[b * KA + r] = i;
    }
}

// context = complement of anchors, ascending token index
__global__ __launch_bounds__(256) void k_ctx(const int* __restrict__ rank, const float* __restrict__ norm2,
                                             int* __restrict__ ctx_idx, float* __restrict__ tnorm2,
                                             float* __restrict__ tcoord) {
    __shared__ int anch[NN];
    int b = blockIdx.x;
    for (int i = threadIdx.x; i < NN; i += 256) anch[i] = (rank[b * NN + i] < KA);
    __syncthreads();
    for (int i = threadIdx.x; i < NN; i += 256) {
        if (!anch[i]) {
            int pos = 0;
            for (int j = 0; j < i; ++j) pos += !anch[j];
            int o = b * MM + pos;
            ctx_idx[o] = i;
            tnorm2[o]  = norm2[b * NN + i];
            tcoord[o * 2 + 0] = (float)(i % SIDE) / (float)SIDE;
            tcoord[o * 2 + 1] = (float)(i / SIDE) / (float)SIDE;
        }
    }
}

__global__ __launch_bounds__(256) void k_gather_bf(const float* __restrict__ x, const int* __restrict__ ctx_idx,
                                                   unsigned short* __restrict__ Tbf) {
    long bm = blockIdx.x; long b = bm / MM;
    const float* src = x + ((long)b * NN + ctx_idx[bm]) * DD;
    unsigned short* dst = Tbf + bm * DD;
    for (int d = threadIdx.x; d < DD; d += 256) dst[d] = f2bf(src[d]);
}

// deterministic centroid init: ctx token at position 3k (stand-in for random permutation)
__global__ __launch_bounds__(256) void k_init_cent(const float* __restrict__ x, const int* __restrict__ ctx_idx,
                                                   const float* __restrict__ tnorm2, const float* __restrict__ tcoord,
                                                   float* __restrict__ cent, unsigned short* __restrict__ Cbf,
                                                   float* __restrict__ cnorm2, float* __restrict__ ccoord) {
    long bk = blockIdx.x; long b = bk / KK; long k = bk % KK;
    long m = b * MM + k * (MM / KK);
    const float* src = x + ((long)b * NN + ctx_idx[m]) * DD;
    for (int d = threadIdx.x; d < DD; d += 256) { float v = src[d]; cent[bk * DD + d] = v; Cbf[bk * DD + d] = f2bf(v); }
    if (threadIdx.x == 0) {
        cnorm2[bk] = tnorm2[m];
        ccoord[bk * 2 + 0] = tcoord[m * 2 + 0];
        ccoord[bk * 2 + 1] = tcoord[m * 2 + 1];
    }
}

// ---------------- WMMA assignment kernel ----------------
// 4 waves/block; wave owns a 16-row strip. Centroid chunks (64 cols x 32 d, bf16)
// streamed global->LDS via async DMA, double-buffered; A fragments software-
// pipelined one chunk ahead so both DMA and A-loads hide under the WMMAs.
__global__ __launch_bounds__(128) void k_assign(const unsigned short* __restrict__ Tbf,
                                                const unsigned short* __restrict__ Cbf,
                                                const float* __restrict__ tnorm2, const float* __restrict__ tcoord,
                                                const float* __restrict__ cnorm2, const float* __restrict__ ccoord,
                                                int* __restrict__ labels) {
    __shared__ __align__(16) unsigned short smem[2][64 * 32];   // double-buffered 4KB chunks
    const int b    = blockIdx.y;
    const int wave = threadIdx.x >> 5;
    const int lane = threadIdx.x & 31;
    const int r0   = blockIdx.x * 64 + wave * 16;
    const int rr   = lane & 15;
    const int h    = lane >> 4;
    const int rbase = r0 + 8 * h;
    const int tcol  = threadIdx.x >> 1;      // 0..63: centroid column within group
    const int thalf = threadIdx.x & 1;

    float tn[8], tx[8], ty[8], best[8]; int bk[8];
    for (int v = 0; v < 8; ++v) {
        int m = b * MM + rbase + v;
        tn[v] = tnorm2[m]; tx[v] = tcoord[m * 2]; ty[v] = tcoord[m * 2 + 1];
        best[v] = 3.4e38f; bk[v] = 0;
    }
    const unsigned short* Tb = Tbf + (long)b * MM * DD;
    const unsigned short* Cb = Cbf + (long)b * KK * DD;
    const unsigned short* Arow = Tb + (long)(r0 + rr) * DD;

    const int T = (KK / 64) * (DD / 32);     // 288 chunks
    // prologue: DMA chunk 0 into buffer 0; preload A fragment for chunk 0
    async_b128x2(Cb + (long)tcol * DD + thalf * 16, &smem[0][tcol * 32 + thalf * 16]);
    bf16x16 a_cur = ld_fragA(Arow, h);

    f32x8 acc[4] = {{0,0,0,0,0,0,0,0},{0,0,0,0,0,0,0,0},{0,0,0,0,0,0,0,0},{0,0,0,0,0,0,0,0}};
    for (int t = 0; t < T; ++t) {
        wait_async0();           // own DMAs for buf[t&1] complete
        __syncthreads();         // all waves' DMAs done; buf[(t+1)&1] released
        const int tnx = (t + 1 < T) ? t + 1 : t;
        if (t + 1 < T) {         // DMA next chunk while computing this one
            const unsigned short* g = Cb + (long)((tnx >> 5) * 64 + tcol) * DD
                                         + (tnx & 31) * 32 + thalf * 16;
            async_b128x2(g, &smem[tnx & 1][tcol * 32 + thalf * 16]);
        }
        bf16x16 a_nxt = ld_fragA(Arow + (tnx & 31) * 32, h);   // A for chunk t+1 in flight

        const unsigned short* buf = smem[t & 1];
        bf16x16 b0 = ld_fragB(buf + (rr) * 32 + h * 16);
        bf16x16 b1 = ld_fragB(buf + (16 + rr) * 32 + h * 16);
        bf16x16 b2 = ld_fragB(buf + (32 + rr) * 32 + h * 16);
        bf16x16 b3 = ld_fragB(buf + (48 + rr) * 32 + h * 16);
        acc[0] = __builtin_amdgcn_wmma_f32_16x16x32_bf16(false, a_cur, false, b0, (short)0, acc[0], false, false);
        acc[1] = __builtin_amdgcn_wmma_f32_16x16x32_bf16(false, a_cur, false, b1, (short)0, acc[1], false, false);
        acc[2] = __builtin_amdgcn_wmma_f32_16x16x32_bf16(false, a_cur, false, b2, (short)0, acc[2], false, false);
        acc[3] = __builtin_amdgcn_wmma_f32_16x16x32_bf16(false, a_cur, false, b3, (short)0, acc[3], false, false);
        a_cur = a_nxt;

        if ((t & 31) == 31) {    // end of a 64-column group: fused distance/argmin epilogue
            const int c0 = (t >> 5) * 64;
            #pragma unroll
            for (int j = 0; j < 4; ++j) {
                int c = c0 + j * 16 + rr;
                float cn2 = cnorm2[b * KK + c];
                float cx = ccoord[(b * KK + c) * 2], cy = ccoord[(b * KK + c) * 2 + 1];
                const float* av = (const float*)&acc[j];
                #pragma unroll
                for (int v = 0; v < 8; ++v) {
                    float fd = sqrtf(fmaxf(tn[v] + cn2 - 2.0f * av[v], 0.0f));
                    float dx = tx[v] - cx, dy = ty[v] - cy;
                    float d  = fd + SW * sqrtf(dx * dx + dy * dy);
                    if (d < best[v]) { best[v] = d; bk[v] = c; }   // cols ascend per lane
                }
                acc[j] = (f32x8){0,0,0,0,0,0,0,0};
            }
        }
    }
    // cross-lane argmin within each 16-lane half (wave32)
    for (int off = 1; off < 16; off <<= 1) {
        #pragma unroll
        for (int v = 0; v < 8; ++v) {
            float ov = __shfl_xor(best[v], off, 32);
            int   ok = __shfl_xor(bk[v],  off, 32);
            if (ov < best[v] || (ov == best[v] && ok < bk[v])) { best[v] = ov; bk[v] = ok; }
        }
    }
    if (rr == 0)
        for (int v = 0; v < 8; ++v) labels[b * MM + rbase + v] = bk[v];
}

// ---------------- segment sums / centroid update ----------------
__global__ __launch_bounds__(256) void k_scatter(const float* __restrict__ x, const int* __restrict__ ctx_idx,
                                                 const int* __restrict__ labels, const float* __restrict__ tcoord,
                                                 float* __restrict__ st, float* __restrict__ cnt, float* __restrict__ sc) {
    long bm = blockIdx.x; long b = bm / MM;
    int lbl = labels[bm];
    const float* src = x + ((long)b * NN + ctx_idx[bm]) * DD;
    float* dst = st + ((long)b * KK + lbl) * DD;
    for (int d = threadIdx.x; d < DD; d += 256) atomicAdd(&dst[d], src[d]);
    if (threadIdx.x == 0) {
        atomicAdd(&cnt[b * KK + lbl], 1.0f);
        atomicAdd(&sc[(b * KK + lbl) * 2 + 0], tcoord[bm * 2 + 0]);
        atomicAdd(&sc[(b * KK + lbl) * 2 + 1], tcoord[bm * 2 + 1]);
    }
}

__global__ __launch_bounds__(256) void k_update(const float* __restrict__ st, const float* __restrict__ cnt,
                                                const float* __restrict__ sc, float* __restrict__ cent,
                                                unsigned short* __restrict__ Cbf, float* __restrict__ cnorm2,
                                                float* __restrict__ ccoord) {
    __shared__ float red[256];
    long bk = blockIdx.x;
    float c = cnt[bk];
    float s = 0.f;
    for (int d = threadIdx.x; d < DD; d += 256) {
        float v = (c > 0.f) ? st[bk * DD + d] / c : cent[bk * DD + d];
        cent[bk * DD + d] = v; Cbf[bk * DD + d] = f2bf(v); s += v * v;
    }
    red[threadIdx.x] = s; __syncthreads();
    for (int o = 128; o > 0; o >>= 1) { if (threadIdx.x < o) red[threadIdx.x] += red[threadIdx.x + o]; __syncthreads(); }
    if (threadIdx.x == 0) {
        cnorm2[bk] = red[0];
        if (c > 0.f) { ccoord[bk * 2] = sc[bk * 2] / c; ccoord[bk * 2 + 1] = sc[bk * 2 + 1] / c; }
    }
}

__global__ __launch_bounds__(256) void k_final_stats(const float* __restrict__ x, const int* __restrict__ ctx_idx,
                                                     const int* __restrict__ labels, const float* __restrict__ cent,
                                                     float* __restrict__ agg, float* __restrict__ idx_sum) {
    long bm = blockIdx.x; long b = bm / MM;
    int lbl = labels[bm]; int idx = ctx_idx[bm];
    const float* src = x + ((long)b * NN + idx) * DD;
    const float* cp  = cent + ((long)b * KK + lbl) * DD;
    float* dst = agg + ((long)b * KK + lbl) * DD;
    for (int d = threadIdx.x; d < DD; d += 256) atomicAdd(&dst[d], src[d] - cp[d]);
    if (threadIdx.x == 0) atomicAdd(&idx_sum[b * KK + lbl], (float)idx);
}

__global__ void k_to_bf16(const float* __restrict__ src, unsigned short* __restrict__ dst, long n) {
    long i = (long)blockIdx.x * blockDim.x + threadIdx.x;
    if (i < n) dst[i] = f2bf(src[i]);
}

// ---------------- WMMA linear kernel: out[r,c] = sum_d A[r,d]*W[c,d] + bias[c] ----------------
__global__ __launch_bounds__(128) void k_linear(const unsigned short* __restrict__ Abf,
                                                const unsigned short* __restrict__ Wbf,
                                                const float* __restrict__ bias, float* __restrict__ out) {
    __shared__ __align__(16) unsigned short smem[2][64 * 32];
    const int wave = threadIdx.x >> 5, lane = threadIdx.x & 31;
    const int r0 = blockIdx.x * 64 + wave * 16;
    const int rr = lane & 15, h = lane >> 4;
    const int tcol = threadIdx.x >> 1, thalf = threadIdx.x & 1;
    const unsigned short* Arow = Abf + (long)(r0 + rr) * DD;

    const int T = (DD / 64) * (DD / 32);     // 512 chunks
    async_b128x2(Wbf + (long)tcol * DD + thalf * 16, &smem[0][tcol * 32 + thalf * 16]);
    bf16x16 a_cur = ld_fragA(Arow, h);

    f32x8 acc[4] = {{0,0,0,0,0,0,0,0},{0,0,0,0,0,0,0,0},{0,0,0,0,0,0,0,0},{0,0,0,0,0,0,0,0}};
    for (int t = 0; t < T; ++t) {
        wait_async0();
        __syncthreads();
        const int tnx = (t + 1 < T) ? t + 1 : t;
        if (t + 1 < T) {
            const unsigned short* g = Wbf + (long)((tnx >> 5) * 64 + tcol) * DD
                                          + (tnx & 31) * 32 + thalf * 16;
            async_b128x2(g, &smem[tnx & 1][tcol * 32 + thalf * 16]);
        }
        bf16x16 a_nxt = ld_fragA(Arow + (tnx & 31) * 32, h);

        const unsigned short* buf = smem[t & 1];
        bf16x16 b0 = ld_fragB(buf + (rr) * 32 + h * 16);
        bf16x16 b1 = ld_fragB(buf + (16 + rr) * 32 + h * 16);
        bf16x16 b2 = ld_fragB(buf + (32 + rr) * 32 + h * 16);
        bf16x16 b3 = ld_fragB(buf + (48 + rr) * 32 + h * 16);
        acc[0] = __builtin_amdgcn_wmma_f32_16x16x32_bf16(false, a_cur, false, b0, (short)0, acc[0], false, false);
        acc[1] = __builtin_amdgcn_wmma_f32_16x16x32_bf16(false, a_cur, false, b1, (short)0, acc[1], false, false);
        acc[2] = __builtin_amdgcn_wmma_f32_16x16x32_bf16(false, a_cur, false, b2, (short)0, acc[2], false, false);
        acc[3] = __builtin_amdgcn_wmma_f32_16x16x32_bf16(false, a_cur, false, b3, (short)0, acc[3], false, false);
        a_cur = a_nxt;

        if ((t & 31) == 31) {
            const int c0 = (t >> 5) * 64;
            #pragma unroll
            for (int j = 0; j < 4; ++j) {
                int c = c0 + j * 16 + rr;
                float bv = bias[c];
                const float* av = (const float*)&acc[j];
                #pragma unroll
                for (int v = 0; v < 8; ++v) out[(long)(r0 + 8 * h + v) * DD + c] = av[v] + bv;
                acc[j] = (f32x8){0,0,0,0,0,0,0,0};
            }
        }
    }
}

// ---------------- output assembly ----------------
__global__ __launch_bounds__(256) void k_coords_nearest(const float* __restrict__ idx_sum, const float* __restrict__ cnt,
                                                        const int* __restrict__ top_idx, int* __restrict__ nearest) {
    int t = blockIdx.x * blockDim.x + threadIdx.x;
    if (t >= BB * KK) return;
    int b = t / KK;
    float c = cnt[t];
    float avg = idx_sum[t] / fmaxf(c, 1.0f);
    float cy = floorf(avg / (float)SIDE) / (float)SIDE;
    float cx = fmodf(avg, (float)SIDE) / (float)SIDE;
    float bd = 3.4e38f; int ba = 0;
    for (int a = 0; a < KA; ++a) {
        int ai = top_idx[b * KA + a];
        float ax = (float)(ai % SIDE) / (float)SIDE;
        float ay = (float)(ai / SIDE) / (float)SIDE;
        float dx = cx - ax, dy = cy - ay, d2 = dx * dx + dy * dy;
        if (d2 < bd) { bd = d2; ba = a; }
    }
    nearest[t] = ba;
}

__global__ __launch_bounds__(256) void k_out_anchor(const float* __restrict__ x, const int* __restrict__ top_idx,
                                                    float* __restrict__ out) {
    long ba = blockIdx.x; long b = ba / KA; long a = ba % KA;
    const float* src = x + ((long)b * NN + top_idx[ba]) * DD;
    float* dst = out + ((long)b * NF + a) * DD;
    for (int d = threadIdx.x; d < DD; d += 256) dst[d] = src[d];
}

__global__ __launch_bounds__(256) void k_out_cent(const float* __restrict__ cent, float* __restrict__ out) {
    long bk = blockIdx.x; long b = bk / KK; long k = bk % KK;
    float* dst = out + ((long)b * NF + KA + k) * DD;
    for (int d = threadIdx.x; d < DD; d += 256) dst[d] = cent[bk * DD + d];
}

__global__ __launch_bounds__(256) void k_inject(const float* __restrict__ agg_lin, const float* __restrict__ cnt,
                                                const int* __restrict__ nearest, const float* __restrict__ rscale,
                                                float* __restrict__ out) {
    long bk = blockIdx.x;
    if (cnt[bk] <= 0.f) return;
    long b = bk / KK;
    float scale = rscale[0];
    float* dst = out + ((long)b * NF + nearest[bk]) * DD;
    const float* src = agg_lin + bk * DD;
    for (int d = threadIdx.x; d < DD; d += 256) atomicAdd(&dst[d], scale * src[d]);
}

__global__ void k_attn(float* __restrict__ attn) {
    int i = blockIdx.x * blockDim.x + threadIdx.x;
    if (i < BB * NF) attn[i] = 1.0f;
}

// ---------------- host launch ----------------
extern "C" void kernel_launch(void* const* d_in, const int* in_sizes, int n_in,
                              void* d_out, int out_size, void* d_ws, size_t ws_size,
                              hipStream_t stream) {
    (void)in_sizes; (void)n_in; (void)out_size; (void)ws_size;
    const float* x     = (const float*)d_in[0];   // [B,N,D]
    const float* W     = (const float*)d_in[1];   // [D,D]
    const float* brs   = (const float*)d_in[2];   // [D]
    const float* rscal = (const float*)d_in[3];   // [1]

    char* ws = (char*)d_ws; size_t off = 0;
    auto take = [&](size_t bytes) -> void* { void* p = ws + off; off = (off + bytes + 255) & ~(size_t)255; return p; };

    float* norm2  = (float*)take((size_t)BB * NN * 4);
    int*   rank   = (int*)  take((size_t)BB * NN * 4);
    int*   topidx = (int*)  take((size_t)BB * KA * 4);
    int*   ctxidx = (int*)  take((size_t)BB * MM * 4);
    float* tnorm2 = (float*)take((size_t)BB * MM * 4);
    float* tcoord = (float*)take((size_t)BB * MM * 8);
    int*   labels = (int*)  take((size_t)BB * MM * 4);
    float* cnorm2 = (float*)take((size_t)BB * KK * 4);
    float* ccoord = (float*)take((size_t)BB * KK * 8);
    float* cnt    = (float*)take((size_t)BB * KK * 4);
    float* sc     = (float*)take((size_t)BB * KK * 8);
    float* idxsum = (float*)take((size_t)BB * KK * 4);
    int*   nearst = (int*)  take((size_t)BB * KK * 4);
    unsigned short* Tbf = (unsigned short*)take((size_t)BB * MM * DD * 2);  // reused for agg-bf16
    float* cent   = (float*)take((size_t)BB * KK * DD * 4);                 // reused for linear output
    unsigned short* Cbf = (unsigned short*)take((size_t)BB * KK * DD * 2);
    float* st     = (float*)take((size_t)BB * KK * DD * 4);                 // reused for agg
    unsigned short* Wbf = (unsigned short*)take((size_t)DD * DD * 2);
    unsigned short* aggbf = Tbf;
    float* agg     = st;
    float* agg_lin = cent;

    float* out  = (float*)d_out;
    float* attn = out + (size_t)BB * NF * DD;

    k_norm2<<<BB * NN, 256, 0, stream>>>(x, norm2);
    k_rank<<<BB, 256, 0, stream>>>(norm2, rank, topidx);
    k_ctx<<<BB, 256, 0, stream>>>(rank, norm2, ctxidx, tnorm2, tcoord);
    k_gather_bf<<<BB * MM, 256, 0, stream>>>(x, ctxidx, Tbf);
    k_init_cent<<<BB * KK, 256, 0, stream>>>(x, ctxidx, tnorm2, tcoord, cent, Cbf, cnorm2, ccoord);
    k_to_bf16<<<(int)(((long)DD * DD + 255) / 256), 256, 0, stream>>>(W, Wbf, (long)DD * DD);

    const long stN = (long)BB * KK * DD;
    for (int it = 0; it < NITER; ++it) {
        k_assign<<<dim3(MM / 64, BB), 128, 0, stream>>>(Tbf, Cbf, tnorm2, tcoord, cnorm2, ccoord, labels);
        k_zero_f32<<<(int)((stN + 255) / 256), 256, 0, stream>>>(st, stN);
        k_zero_f32<<<(BB * KK + 255) / 256, 256, 0, stream>>>(cnt, BB * KK);
        k_zero_f32<<<(2 * BB * KK + 255) / 256, 256, 0, stream>>>(sc, 2 * BB * KK);
        k_scatter<<<BB * MM, 256, 0, stream>>>(x, ctxidx, labels, tcoord, st, cnt, sc);
        k_update<<<BB * KK, 256, 0, stream>>>(st, cnt, sc, cent, Cbf, cnorm2, ccoord);
    }

    // final residual aggregation with last labels / updated centroids
    k_zero_f32<<<(int)((stN + 255) / 256), 256, 0, stream>>>(agg, stN);
    k_zero_f32<<<(BB * KK + 255) / 256, 256, 0, stream>>>(idxsum, BB * KK);
    k_final_stats<<<BB * MM, 256, 0, stream>>>(x, ctxidx, labels, cent, agg, idxsum);

    // output centroids (frees `cent` buffer for linear output)
    k_out_anchor<<<BB * KA, 256, 0, stream>>>(x, topidx, out);
    k_out_cent<<<BB * KK, 256, 0, stream>>>(cent, out);
    k_attn<<<(BB * NF + 255) / 256, 256, 0, stream>>>(attn);

    // linear: agg @ W^T + b  (bf16 WMMA GEMM over [B*K, D] x [D, D])
    k_to_bf16<<<(int)((stN + 255) / 256), 256, 0, stream>>>(agg, aggbf, stN);
    k_linear<<<BB * KK / 64, 128, 0, stream>>>(aggbf, Wbf, brs, agg_lin);

    k_coords_nearest<<<(BB * KK + 255) / 256, 256, 0, stream>>>(idxsum, cnt, topidx, nearst);
    k_inject<<<BB * KK, 256, 0, stream>>>(agg_lin, cnt, nearst, rscal, out);
}
